// MoELayer_50319836840658
// MI455X (gfx1250) — matrix operand order repaired and checked by reading, not compiled, not run on hardware
//
#include <hip/hip_runtime.h>
#include <hip/hip_bf16.h>
#include <math.h>

// ---------------- problem constants ----------------
#define B_   2
#define S_   2048
#define D_   1024
#define E_   8
#define K_   2
#define H_   4096
#define NTOK (B_ * S_)      // 4096 tokens
#define NASG (NTOK * K_)    // 8192 (token, expert) assignments
#define MAX_TILES 264       // sum_e ceil(cnt_e/32) <= 8192/32 + 7

typedef __attribute__((ext_vector_type(16))) __bf16 v16bf;
typedef __attribute__((ext_vector_type(8)))  float  v8f;

union FragBF {
    v16bf v;
    uint4 q[2];
    __bf16 h[16];
};

// ---------------- small helpers ----------------
__global__ void zero_f32_kernel(float* p, int n) {
    for (int i = blockIdx.x * blockDim.x + threadIdx.x; i < n;
         i += gridDim.x * blockDim.x)
        p[i] = 0.0f;
}

__global__ void init_small_kernel(float* meSum, int* cnt) {
    int t = threadIdx.x;
    if (t < E_) { meSum[t] = 0.0f; cnt[t] = 0; }
}

__global__ void cvt_bf16_kernel(const float* __restrict__ in, __bf16* __restrict__ out, int n) {
    for (int i = blockIdx.x * blockDim.x + threadIdx.x; i < n;
         i += gridDim.x * blockDim.x)
        out[i] = (__bf16)in[i];
}

// fp32 -> bf16 transposing convert: out[e][c][r] = (bf16) in[e][r][c]
// in is [E][R][C] row-major. grid = (C/32, R/32, E), block = (32, 8)
__global__ void cvt_tr_bf16_kernel(const float* __restrict__ in, __bf16* __restrict__ out,
                                   int R, int C) {
    __shared__ float tile[32][33];
    int e  = blockIdx.z;
    int r0 = blockIdx.y * 32;
    int c0 = blockIdx.x * 32;
    const float* ine = in + (size_t)e * R * C;
    __bf16* oute     = out + (size_t)e * R * C;
#pragma unroll
    for (int i = 0; i < 4; i++) {
        int r = r0 + threadIdx.y + i * 8;
        tile[threadIdx.y + i * 8][threadIdx.x] = ine[(size_t)r * C + c0 + threadIdx.x];
    }
    __syncthreads();
#pragma unroll
    for (int i = 0; i < 4; i++) {
        int c = c0 + threadIdx.y + i * 8;
        oute[(size_t)c * R + r0 + threadIdx.x] = (__bf16)tile[threadIdx.x][threadIdx.y + i * 8];
    }
}

// ---------------- gating: one wave per token ----------------
__global__ void gate_kernel(const float* __restrict__ x, const float* __restrict__ Wg,
                            const float* __restrict__ bg,
                            int* __restrict__ selOut,
                            float* __restrict__ meSum, int* __restrict__ cnt,
                            int* __restrict__ topPack, float* __restrict__ topP1) {
    int tok  = (blockIdx.x * blockDim.x + threadIdx.x) >> 5;
    int lane = threadIdx.x & 31;
    if (tok >= NTOK) return;

    const float* xr = x + (size_t)tok * D_;
    float acc[E_];
#pragma unroll
    for (int e = 0; e < E_; e++) acc[e] = 0.0f;

    for (int d = lane; d < D_; d += 32) {
        float xv = xr[d];
        const float4* wr = (const float4*)(Wg + (size_t)d * E_);
        float4 w0 = wr[0], w1 = wr[1];
        acc[0] += xv * w0.x; acc[1] += xv * w0.y; acc[2] += xv * w0.z; acc[3] += xv * w0.w;
        acc[4] += xv * w1.x; acc[5] += xv * w1.y; acc[6] += xv * w1.z; acc[7] += xv * w1.w;
    }
#pragma unroll
    for (int e = 0; e < E_; e++)
        for (int off = 16; off; off >>= 1)
            acc[e] += __shfl_xor(acc[e], off, 32);

    if (lane == 0) {
        float lg[E_];
#pragma unroll
        for (int e = 0; e < E_; e++) lg[e] = acc[e] + bg[e];

        float mx = lg[0];
#pragma unroll
        for (int e = 1; e < E_; e++) mx = fmaxf(mx, lg[e]);
        float se = 0.0f, pr[E_];
#pragma unroll
        for (int e = 0; e < E_; e++) { pr[e] = __expf(lg[e] - mx); se += pr[e]; }
        float inv = 1.0f / se;
#pragma unroll
        for (int e = 0; e < E_; e++) atomicAdd(&meSum[e], pr[e] * inv);

        int e1 = 0;
#pragma unroll
        for (int e = 1; e < E_; e++) if (lg[e] > lg[e1]) e1 = e;
        int e2 = -1;
#pragma unroll
        for (int e = 0; e < E_; e++)
            if (e != e1 && (e2 < 0 || lg[e] > lg[e2])) e2 = e;

        float p1 = 1.0f / (1.0f + __expf(lg[e2] - lg[e1]));

        selOut[tok * K_ + 0] = e1;
        selOut[tok * K_ + 1] = e2;
        topPack[tok] = e1 | (e2 << 8);
        topP1[tok]   = p1;
        atomicAdd(&cnt[e1], 1);
        atomicAdd(&cnt[e2], 1);
    }
}

// ---------------- scan: offsets, 32-row tile table, aux loss ----------------
__global__ void scan_kernel(const int* __restrict__ cnt, const float* __restrict__ meSum,
                            int* __restrict__ offs, int* __restrict__ cursor,
                            int* __restrict__ tiles, float* __restrict__ auxOut) {
    if (threadIdx.x == 0 && blockIdx.x == 0) {
        int o = 0, t = 0;
        float aux = 0.0f;
        for (int e = 0; e < E_; e++) {
            offs[e] = o; cursor[e] = o; tiles[e] = t;
            o += cnt[e];
            t += (cnt[e] + 31) >> 5;
            float me = meSum[e] / (float)NTOK;
            float ce = (float)cnt[e] / (float)NTOK;
            aux += me * ce;
        }
        offs[E_] = o; tiles[E_] = t;
        *auxOut = (float)E_ * aux;
    }
}

// ---------------- scatter tokens into per-expert lists ----------------
__global__ void scatter_kernel(const int* __restrict__ topPack, const float* __restrict__ topP1,
                               int* __restrict__ cursor,
                               int* __restrict__ tokList, float* __restrict__ wList) {
    int t = blockIdx.x * blockDim.x + threadIdx.x;
    if (t >= NTOK) return;
    int pk = topPack[t];
    int e1 = pk & 0xff, e2 = (pk >> 8) & 0xff;
    float p1 = topP1[t];
    int s1 = atomicAdd(&cursor[e1], 1);
    tokList[s1] = t; wList[s1] = p1;
    int s2 = atomicAdd(&cursor[e2], 1);
    tokList[s2] = t; wList[s2] = 1.0f - p1;
}

// ---------------- FFN1: h = gelu(x_sel @ W1[e] + b1[e]) ----------------
// Weights pre-transposed: w1t[e][n][k], k contiguous (k = D dim, n = H dim).
// grid = (H/256, MAX_TILES), block = 128 (4 waves); wave computes 32(M) x 64(N):
// 2 A row-groups x 4 N subtiles; fits the 128-VGPR budget with zero spills.
__global__ void ffn1_kernel(const __bf16* __restrict__ xb, const __bf16* __restrict__ w1t,
                            const float* __restrict__ b1,
                            const int* __restrict__ offs, const int* __restrict__ tiles,
                            const int* __restrict__ tokList,
                            __bf16* __restrict__ hb) {
    int tileM = blockIdx.y;
    if (tileM >= tiles[E_]) return;
    int e = 0;
    while (tileM >= tiles[e + 1]) e++;
    int lt  = tileM - tiles[e];
    int off = offs[e];
    int cnt = offs[e + 1] - off;

    int wv    = threadIdx.x >> 5;
    int lane  = threadIdx.x & 31;
    int half  = lane >> 4;        // K-half within wave
    int lr    = lane & 15;        // row (A) / column (B,C)
    int nbase = blockIdx.x * 256 + wv * 64;
    int akoff = half * 8;

    int r0 = lt * 32 + lr;
    int r1 = r0 + 16;
    int tok0 = tokList[off + (r0 < cnt ? r0 : 0)];
    int tok1 = tokList[off + (r1 < cnt ? r1 : 0)];
    const __bf16* arow0 = xb + (size_t)tok0 * D_;
    const __bf16* arow1 = xb + (size_t)tok1 * D_;
    const __bf16* bbase = w1t + (size_t)e * D_ * H_;

    v8f acc[2][4];
    v8f zero = {0.f, 0.f, 0.f, 0.f, 0.f, 0.f, 0.f, 0.f};
#pragma unroll
    for (int g = 0; g < 2; g++)
#pragma unroll
        for (int t = 0; t < 4; t++) acc[g][t] = zero;

    for (int k0 = 0; k0 < D_; k0 += 32) {
        FragBF A0, A1;
        A0.q[0] = *(const uint4*)(arow0 + k0 + akoff);
        A0.q[1] = *(const uint4*)(arow0 + k0 + akoff + 16);
        A1.q[0] = *(const uint4*)(arow1 + k0 + akoff);
        A1.q[1] = *(const uint4*)(arow1 + k0 + akoff + 16);
        __builtin_prefetch(arow0 + k0 + 64, 0, 1);          // global_prefetch_b8
#pragma unroll
        for (int t = 0; t < 4; t++) {
            int ncol = nbase + t * 16 + lr;
            const __bf16* bp = bbase + (size_t)ncol * D_ + k0 + half * 8;
            FragBF Bf;
            Bf.q[0] = *(const uint4*)(bp);
            Bf.q[1] = *(const uint4*)(bp + 16);
            __builtin_prefetch(bp + 32, 0, 1);
            acc[0][t] = __builtin_amdgcn_wmma_f32_16x16x32_bf16(
                false, A0.v, false, Bf.v, (short)0, acc[0][t], false, false);
            acc[1][t] = __builtin_amdgcn_wmma_f32_16x16x32_bf16(
                false, A1.v, false, Bf.v, (short)0, acc[1][t], false, false);
        }
    }

#pragma unroll
    for (int g = 0; g < 2; g++)
#pragma unroll
        for (int v = 0; v < 8; v++) {
            int m   = g * 16 + v + half * 8;
            int idx = lt * 32 + m;
            if (idx < cnt) {
                size_t row = (size_t)(off + idx) * H_;
#pragma unroll
                for (int t = 0; t < 4; t++) {
                    int ncol  = nbase + t * 16 + lr;
                    float val = acc[g][t][v] + b1[e * H_ + ncol];
                    val = 0.5f * val * (1.0f + erff(val * 0.70710678f));   // exact GELU
                    hb[row + ncol] = (__bf16)val;
                }
            }
        }
}

// ---------------- FFN2: out += w * (h @ W2[e] + b2[e]) ----------------
// Weights pre-transposed: w2t[e][n][k], k contiguous (k = H dim, n = D dim).
// grid = (D/256, MAX_TILES), block = 128.
__global__ void ffn2_kernel(const __bf16* __restrict__ hb, const __bf16* __restrict__ w2t,
                            const float* __restrict__ b2,
                            const int* __restrict__ offs, const int* __restrict__ tiles,
                            const int* __restrict__ tokList, const float* __restrict__ wList,
                            float* __restrict__ out) {
    int tileM = blockIdx.y;
    if (tileM >= tiles[E_]) return;
    int e = 0;
    while (tileM >= tiles[e + 1]) e++;
    int lt  = tileM - tiles[e];
    int off = offs[e];
    int cnt = offs[e + 1] - off;

    int wv    = threadIdx.x >> 5;
    int lane  = threadIdx.x & 31;
    int half  = lane >> 4;
    int lr    = lane & 15;
    int nbase = blockIdx.x * 256 + wv * 64;
    int akoff = half * 8;

    int r0 = lt * 32 + lr;
    int r1 = r0 + 16;
    int slot0 = off + (r0 < cnt ? r0 : 0);
    int slot1 = off + (r1 < cnt ? r1 : 0);
    const __bf16* arow0 = hb + (size_t)slot0 * H_;
    const __bf16* arow1 = hb + (size_t)slot1 * H_;
    const __bf16* bbase = w2t + (size_t)e * H_ * D_;

    v8f acc[2][4];
    v8f zero = {0.f, 0.f, 0.f, 0.f, 0.f, 0.f, 0.f, 0.f};
#pragma unroll
    for (int g = 0; g < 2; g++)
#pragma unroll
        for (int t = 0; t < 4; t++) acc[g][t] = zero;

    for (int k0 = 0; k0 < H_; k0 += 32) {
        FragBF A0, A1;
        A0.q[0] = *(const uint4*)(arow0 + k0 + akoff);
        A0.q[1] = *(const uint4*)(arow0 + k0 + akoff + 16);
        A1.q[0] = *(const uint4*)(arow1 + k0 + akoff);
        A1.q[1] = *(const uint4*)(arow1 + k0 + akoff + 16);
        __builtin_prefetch(arow0 + k0 + 64, 0, 1);
#pragma unroll
        for (int t = 0; t < 4; t++) {
            int ncol = nbase + t * 16 + lr;
            const __bf16* bp = bbase + (size_t)ncol * H_ + k0 + half * 8;
            FragBF Bf;
            Bf.q[0] = *(const uint4*)(bp);
            Bf.q[1] = *(const uint4*)(bp + 16);
            __builtin_prefetch(bp + 32, 0, 1);
            acc[0][t] = __builtin_amdgcn_wmma_f32_16x16x32_bf16(
                false, A0.v, false, Bf.v, (short)0, acc[0][t], false, false);
            acc[1][t] = __builtin_amdgcn_wmma_f32_16x16x32_bf16(
                false, A1.v, false, Bf.v, (short)0, acc[1][t], false, false);
        }
    }

#pragma unroll
    for (int g = 0; g < 2; g++)
#pragma unroll
        for (int v = 0; v < 8; v++) {
            int m   = g * 16 + v + half * 8;
            int idx = lt * 32 + m;
            if (idx < cnt) {
                int   sm  = off + idx;
                int   tok = tokList[sm];
                float w   = wList[sm];
#pragma unroll
                for (int t = 0; t < 4; t++) {
                    int ncol  = nbase + t * 16 + lr;
                    float val = acc[g][t][v] + b2[e * D_ + ncol];
                    unsafeAtomicAdd(&out[(size_t)tok * D_ + ncol], w * val);
                }
            }
        }
}

// ---------------- host-side orchestration ----------------
extern "C" void kernel_launch(void* const* d_in, const int* in_sizes, int n_in,
                              void* d_out, int out_size, void* d_ws, size_t ws_size,
                              hipStream_t stream) {
    const float* x  = (const float*)d_in[0];
    const float* Wg = (const float*)d_in[1];
    const float* bg = (const float*)d_in[2];
    const float* W1 = (const float*)d_in[3];
    const float* b1 = (const float*)d_in[4];
    const float* W2 = (const float*)d_in[5];
    const float* b2 = (const float*)d_in[6];

    float* out    = (float*)d_out;
    int*   selOut = (int*)(out + (size_t)NTOK * D_);
    float* auxOut = out + (size_t)NTOK * D_ + NASG;

    // workspace carve-up (all large arrays 16B-aligned)
    char* w = (char*)d_ws;
    __bf16* xb  = (__bf16*)w; w += (size_t)NTOK * D_ * 2;
    __bf16* w1t = (__bf16*)w; w += (size_t)E_ * D_ * H_ * 2;   // [E][H][D] transposed
    __bf16* w2t = (__bf16*)w; w += (size_t)E_ * H_ * D_ * 2;   // [E][D][H] transposed
    __bf16* hb  = (__bf16*)w; w += (size_t)NASG * H_ * 2;
    int*   tokList = (int*)w;   w += (size_t)NASG * 4;
    float* wList   = (float*)w; w += (size_t)NASG * 4;
    int*   topPack = (int*)w;   w += (size_t)NTOK * 4;
    float* topP1   = (float*)w; w += (size_t)NTOK * 4;
    float* meSum   = (float*)w; w += E_ * 4;
    int*   cnt     = (int*)w;   w += E_ * 4;
    int*   offs    = (int*)w;   w += (E_ + 1) * 4;
    int*   cursor  = (int*)w;   w += E_ * 4;
    int*   tiles   = (int*)w;   w += (E_ + 1) * 4;

    init_small_kernel<<<1, 64, 0, stream>>>(meSum, cnt);
    zero_f32_kernel<<<2048, 256, 0, stream>>>(out, NTOK * D_);

    cvt_bf16_kernel<<<1024, 256, 0, stream>>>(x, xb, NTOK * D_);
    // W1 [E][D][H] -> w1t [E][H][D]
    cvt_tr_bf16_kernel<<<dim3(H_ / 32, D_ / 32, E_), dim3(32, 8), 0, stream>>>(W1, w1t, D_, H_);
    // W2 [E][H][D] -> w2t [E][D][H]
    cvt_tr_bf16_kernel<<<dim3(D_ / 32, H_ / 32, E_), dim3(32, 8), 0, stream>>>(W2, w2t, H_, D_);

    gate_kernel<<<NTOK / 8, 256, 0, stream>>>(x, Wg, bg, selOut, meSum, cnt, topPack, topP1);
    scan_kernel<<<1, 1, 0, stream>>>(cnt, meSum, offs, cursor, tiles, auxOut);
    scatter_kernel<<<NTOK / 256, 256, 0, stream>>>(topPack, topP1, cursor, tokList, wList);

    ffn1_kernel<<<dim3(H_ / 256, MAX_TILES), 128, 0, stream>>>(
        xb, w1t, b1, offs, tiles, tokList, hb);
    ffn2_kernel<<<dim3(D_ / 256, MAX_TILES), 128, 0, stream>>>(
        hb, w2t, b2, offs, tiles, tokList, wList, out);
}
